// RegularAttention_9148280341032
// MI455X (gfx1250) — matrix operand order repaired and checked by reading, not compiled, not run on hardware
//
#include <hip/hip_runtime.h>

typedef __attribute__((ext_vector_type(16))) _Float16 v16h;
typedef __attribute__((ext_vector_type(2)))  _Float16 h2;
typedef __attribute__((ext_vector_type(8)))  float    v8f;
typedef __attribute__((ext_vector_type(4)))  unsigned int u32x4;
typedef __attribute__((ext_vector_type(8)))  int          i32x8;
typedef __attribute__((ext_vector_type(4)))  int          i32x4;

static constexpr int B_ = 2;
static constexpr int H_ = 12;
static constexpr int S_ = 2048;
static constexpr int D_ = 64;
static constexpr int W_ = 256;

#if __has_builtin(__builtin_amdgcn_tensor_load_to_lds)
#define HAS_TDM 1
#else
#define HAS_TDM 0
#endif

// ---- LDS layout (single arena so TDM byte offsets are well defined) ----
// Padded strides give bank-conflict-free b128 fragment reads:
//   Kh  row stride 72 halfs (144B): bank index 36*n  -> 16 distinct banks
//   VhT/P row stride 40 halfs (80B): bank index 20*n -> 16 distinct banks
static constexpr int KH_STR = 72;   // halfs per Kh row   [32][72]
static constexpr int VT_STR = 40;   // halfs per VhT row  [64][40] (d-major)
static constexpr int P_STR  = 40;   // halfs per P row    [16][40] per wave

static constexpr int KF32_OFF = 0;        // 2 x (32x64 f32) = 16384 B
static constexpr int VF32_OFF = 16384;    // 2 x (32x64 f32) = 16384 B
static constexpr int KH_OFF   = 32768;    // 32*72*2 = 4608 B
static constexpr int VT_OFF   = 37376;    // 64*40*2 = 5120 B
static constexpr int P_OFF    = 42496;    // 4*16*40*2 = 5120 B
static constexpr int SMEM_BYTES = 47616;

// Issue one TDM 2D tile load: 32 rows x 64 fp32 cols, row stride 64 elements.
// D# packing per CDNA5 ISA ch.8.
__device__ __forceinline__ void tdm_load_tile(unsigned lds_addr, const float* gsrc) {
#if HAS_TDM
  unsigned long long ga = (unsigned long long)(uintptr_t)gsrc;
  u32x4 g0;
  g0[0] = 1u;                                            // count=1, user mode
  g0[1] = lds_addr;                                      // LDS byte address
  g0[2] = (unsigned)ga;                                  // global_addr[31:0]
  g0[3] = ((unsigned)(ga >> 32) & 0x01FFFFFFu)           // global_addr[56:32]
          | 0x80000000u;                                 // type=2 ("image")
  i32x8 g1;
  g1[0] = 0x00020000;          // workgroup_mask=0, data_size=2 (4 bytes)
  g1[1] = (int)(64u << 16);    // tensor_dim0 = 64
  g1[2] = (int)(32u << 16);    // tensor_dim1 = 32
  g1[3] = (int)(64u << 16);    // tile_dim0 = 64
  g1[4] = 32;                  // tile_dim1 = 32, tile_dim2 = 0
  g1[5] = 64;                  // tensor_dim0_stride = 64
  g1[6] = 0;
  g1[7] = 0;
  i32x4 z4 = {0, 0, 0, 0};
#if defined(__clang_major__) && (__clang_major__ >= 23)
  i32x8 z8 = {0, 0, 0, 0, 0, 0, 0, 0};
  __builtin_amdgcn_tensor_load_to_lds(g0, g1, z4, z4, z8, 0);
#else
  __builtin_amdgcn_tensor_load_to_lds(g0, g1, z4, z4, 0);
#endif
#else
  (void)lds_addr; (void)gsrc;
#endif
}

__device__ __forceinline__ void wait_tensorcnt_2() {
#if __has_builtin(__builtin_amdgcn_s_wait_tensorcnt)
  __builtin_amdgcn_s_wait_tensorcnt(2);
#else
  asm volatile("s_wait_tensorcnt 0x2" ::: "memory");
#endif
}
__device__ __forceinline__ void wait_tensorcnt_0() {
#if __has_builtin(__builtin_amdgcn_s_wait_tensorcnt)
  __builtin_amdgcn_s_wait_tensorcnt(0);
#else
  asm volatile("s_wait_tensorcnt 0x0" ::: "memory");
#endif
}

// Flash-style sliding-window attention.
// Block = 4 wave32s = 64 query rows of one (b,h); K/V tiles staged per block
// by the Tensor Data Mover (double buffered), converted once to f16 in LDS
// (V stored transposed so P@V B-fragments are contiguous b128 reads).
__global__ __launch_bounds__(128, 1) void swa_flash_wmma_tdm(
    const float* __restrict__ Q,
    const float* __restrict__ K,
    const float* __restrict__ V,
    float* __restrict__ O)
{
  const int bh   = blockIdx.y;
  const int tid  = threadIdx.x;
  const int wave = tid >> 5;
  const int lane = tid & 31;
  const int n    = lane & 15;   // column / row-in-tile index
  const int hh   = lane >> 4;   // which half of the wave

  const int qbase = blockIdx.x * 64;
  const int qrow0 = qbase + wave * 16;

  const float* Qb = Q + (size_t)bh * S_ * D_;
  const float* Kb = K + (size_t)bh * S_ * D_;
  const float* Vb = V + (size_t)bh * S_ * D_;
  float*       Ob = O + (size_t)bh * S_ * D_;

  __shared__ __align__(16) unsigned char smem[SMEM_BYTES];
  const unsigned lds_base = (unsigned)(uintptr_t)&smem[0];

  _Float16* Kh  = (_Float16*)(smem + KH_OFF);                 // [32][KH_STR]
  _Float16* VhT = (_Float16*)(smem + VT_OFF);                 // [64][VT_STR] (d-major)
  _Float16* Pb  = (_Float16*)(smem + P_OFF) + wave * 16 * P_STR;

  // ---- Load Q tile (16x64 fp32) as two f16 A-fragments (contraction 0..31 / 32..63)
  v16h aQ0, aQ1;
  {
    const float* qr = Qb + (size_t)(qrow0 + n) * D_;
#pragma unroll
    for (int g = 0; g < 8; ++g) {
      const int d0 = 16 * hh + 2 * g;
      float2 f0 = *(const float2*)(qr + d0);
      float2 f1 = *(const float2*)(qr + 32 + d0);
      aQ0[2*g] = (_Float16)f0.x;  aQ0[2*g+1] = (_Float16)f0.y;
      aQ1[2*g] = (_Float16)f1.x;  aQ1[2*g+1] = (_Float16)f1.y;
    }
  }

  v8f acc[4] = {};
  float mrow[8], lrow[8];
#pragma unroll
  for (int g = 0; g < 8; ++g) { mrow[g] = -1.0e30f; lrow[g] = 0.0f; }

  // Union key range of the whole block: (qbase - W, qbase + 63]
  int kbegin = qbase - W_ + 1;
  if (kbegin < 0) kbegin = 0;
  kbegin &= ~31;
  const int kend    = qbase + 64;                 // multiple of 32, <= S_
  const int nchunks = (kend - kbegin) >> 5;       // <= 10

#if HAS_TDM
  if (wave == 0) {   // prologue: DMA chunk 0 into buffer 0
    tdm_load_tile(lds_base + KF32_OFF, Kb + (size_t)kbegin * D_);
    tdm_load_tile(lds_base + VF32_OFF, Vb + (size_t)kbegin * D_);
  }
#endif

  for (int ic = 0; ic < nchunks; ++ic) {
    const int kb  = kbegin + ic * 32;
    const int buf = ic & 1;

#if HAS_TDM
    if (wave == 0) {
      if (ic + 1 < nchunks) {
        const int kn = kb + 32;
        tdm_load_tile(lds_base + KF32_OFF + (buf ^ 1) * 8192, Kb + (size_t)kn * D_);
        tdm_load_tile(lds_base + VF32_OFF + (buf ^ 1) * 8192, Vb + (size_t)kn * D_);
        wait_tensorcnt_2();   // TDM in-order: chunk ic's two loads are done
      } else {
        wait_tensorcnt_0();
      }
    }
#else
    {  // fallback staging: cooperative vector loads
      float4*       kd = (float4*)(smem + KF32_OFF + buf * 8192);
      float4*       vd = (float4*)(smem + VF32_OFF + buf * 8192);
      const float4* ks = (const float4*)(Kb + (size_t)kb * D_);
      const float4* vs = (const float4*)(Vb + (size_t)kb * D_);
#pragma unroll
      for (int e = 0; e < 4; ++e) { kd[tid + 128 * e] = ks[tid + 128 * e];
                                    vd[tid + 128 * e] = vs[tid + 128 * e]; }
    }
#endif
    __syncthreads();

    // ---- one f32->f16 conversion per element, shared by all 4 waves.
    // K kept row-major (padded); V written transposed (d-major, padded).
    {
      const float2* ks = (const float2*)(smem + KF32_OFF + buf * 8192);
      const float2* vs = (const float2*)(smem + VF32_OFF + buf * 8192);
#pragma unroll
      for (int e = 0; e < 8; ++e) {
        const int i2 = tid + 128 * e;        // float2 index: conflict-free banks
        const int r  = (2 * i2) >> 6;        // tile row 0..31
        const int d  = (2 * i2) & 63;        // even column 0..62
        float2 fk = ks[i2];
        float2 fv = vs[i2];
        h2 hk; hk[0] = (_Float16)fk.x; hk[1] = (_Float16)fk.y;
        *(h2*)(Kh + r * KH_STR + d) = hk;
        VhT[(d)     * VT_STR + r] = (_Float16)fv.x;
        VhT[(d + 1) * VT_STR + r] = (_Float16)fv.y;
      }
    }
    __syncthreads();

    // Does this wave's window intersect this chunk?
    const bool active = (kb <= qrow0 + 15) && (kb + 31 >= qrow0 - (W_ - 1));
    if (active) {
      // ---- scores: two 16x16 tiles, contraction over D=64 (2 WMMAs each)
      v8f sc[2];
#pragma unroll
      for (int sub = 0; sub < 2; ++sub) {
        const _Float16* kr = Kh + (sub * 16 + n) * KH_STR;
        v16h bK0, bK1;
#pragma unroll
        for (int g = 0; g < 8; ++g) {
          const int d0 = 16 * hh + 2 * g;
          h2 a = *(const h2*)(kr + d0);
          h2 b = *(const h2*)(kr + 32 + d0);
          bK0[2*g] = a[0]; bK0[2*g+1] = a[1];
          bK1[2*g] = b[0]; bK1[2*g+1] = b[1];
        }
        v8f s = {};
        s = __builtin_amdgcn_wmma_f32_16x16x32_f16(false, aQ0, false, bK0,
                                                   (short)0, s, false, false);
        s = __builtin_amdgcn_wmma_f32_16x16x32_f16(false, aQ1, false, bK1,
                                                   (short)0, s, false, false);
        sc[sub] = s;
      }

      // ---- sliding-window mask (skipped for interior chunks: wave-uniform)
      const bool full = (kb + 31 <= qrow0) && (kb >= qrow0 + 16 - W_);
      float s0v[8], s1v[8];
      if (full) {
#pragma unroll
        for (int g = 0; g < 8; ++g) { s0v[g] = sc[0][g]; s1v[g] = sc[1][g]; }
      } else {
#pragma unroll
        for (int g = 0; g < 8; ++g) {
          const int qi = qrow0 + g + 8 * hh;
          // j in (qi-W, qi]  <=>  (unsigned)(qi - j) < W
          s0v[g] = ((unsigned)(qi - (kb + n))      < (unsigned)W_) ? sc[0][g] : -3.0e38f;
          s1v[g] = ((unsigned)(qi - (kb + 16 + n)) < (unsigned)W_) ? sc[1][g] : -3.0e38f;
        }
      }

      // ---- online softmax update
      float scale[8];
#pragma unroll
      for (int g = 0; g < 8; ++g) {
        float tmax = fmaxf(s0v[g], s1v[g]);
#pragma unroll
        for (int off = 1; off < 16; off <<= 1)
          tmax = fmaxf(tmax, __shfl_xor(tmax, off, 32));
        const float newm = fmaxf(mrow[g], tmax);
        scale[g] = __expf(mrow[g] - newm);
        const float p0 = __expf(s0v[g] - newm);
        const float p1 = __expf(s1v[g] - newm);
        lrow[g] = lrow[g] * scale[g] + p0 + p1;
        mrow[g] = newm;
        Pb[(g + 8 * hh) * P_STR + n]      = (_Float16)p0;
        Pb[(g + 8 * hh) * P_STR + 16 + n] = (_Float16)p1;
      }

      // Same-wave LDS ops are in-order; compiler barrier + explicit wait.
      asm volatile("s_wait_dscnt 0" ::: "memory");

#pragma unroll
      for (int t = 0; t < 4; ++t)
#pragma unroll
        for (int g = 0; g < 8; ++g)
          acc[t][g] *= scale[g];

      // ---- reload P (16x32) in A-fragment layout (two b128 reads per lane)
      v16h aP;
#pragma unroll
      for (int g = 0; g < 8; ++g) {
        const int kk = 16 * hh + 2 * g;
        h2 t2 = *(const h2*)(Pb + n * P_STR + kk);
        aP[2*g] = t2[0];  aP[2*g+1] = t2[1];
      }

      // ---- O += P @ V: B-fragment is 16 contiguous halfs of transposed V
#pragma unroll
      for (int t = 0; t < 4; ++t) {
        const _Float16* vr = VhT + (t * 16 + n) * VT_STR + 16 * hh;
        v16h bV;
#pragma unroll
        for (int e = 0; e < 16; ++e) bV[e] = vr[e];
        acc[t] = __builtin_amdgcn_wmma_f32_16x16x32_f16(false, aP, false, bV,
                                                        (short)0, acc[t], false, false);
      }
    }
  }

  // ---- finalize: divide by row sums and store
#pragma unroll
  for (int g = 0; g < 8; ++g) {
    float lsum = lrow[g];
#pragma unroll
    for (int off = 1; off < 16; off <<= 1)
      lsum += __shfl_xor(lsum, off, 32);
    const float inv = 1.0f / lsum;
    float* orow = Ob + (size_t)(qrow0 + g + 8 * hh) * D_ + n;
#pragma unroll
    for (int t = 0; t < 4; ++t)
      orow[t * 16] = acc[t][g] * inv;
  }
}

extern "C" void kernel_launch(void* const* d_in, const int* in_sizes, int n_in,
                              void* d_out, int out_size, void* d_ws, size_t ws_size,
                              hipStream_t stream) {
  (void)in_sizes; (void)n_in; (void)d_ws; (void)ws_size; (void)out_size;
  const float* q = (const float*)d_in[0];
  const float* k = (const float*)d_in[1];
  const float* v = (const float*)d_in[2];
  // d_in[3] is the mask; the sliding-window predicate is computed analytically.
  float* out = (float*)d_out;

  dim3 grid(S_ / 64, B_ * H_);
  dim3 block(128);
  hipLaunchKernelGGL(swa_flash_wmma_tdm, grid, block, 0, stream, q, k, v, out);
}
